// MambaRef_6597069767179
// MI455X (gfx1250) — compile-verified, matrix operand order checked by reading, not run on hardware
//
#include <hip/hip_runtime.h>
#include <math.h>

typedef __attribute__((ext_vector_type(16))) __bf16 v16bf;
typedef __attribute__((ext_vector_type(8)))  float  v8f;

#define B_  2
#define L_  1024
#define DM  1024
#define DI  2048
#define DS  16
#define DC  4
#define BL  (B_*L_)   // 2048 rows

// workspace layout (bytes), all offsets 256B-aligned
#define OFF_XZ    0u                                   // BL x 4096 f32 = 32 MB  (x_in | z)
#define OFF_XC    (OFF_XZ   + (size_t)BL*4096u*4u)     // BL x 2048 f32 = 16 MB
#define OFF_DBC   (OFF_XC   + (size_t)BL*2048u*4u)     // BL x 33 f32 ~ 270 KB
#define OFF_YBF   (OFF_DBC  + 272384u)                 // BL x 2048 bf16 = 8 MB
#define OFF_XBF   (OFF_YBF  + (size_t)BL*2048u*2u)     // BL x 1024 bf16 = 4 MB
#define OFF_WINT  (OFF_XBF  + (size_t)BL*1024u*2u)     // 4096 x 1024 bf16 = 8 MB
#define OFF_WOUTT (OFF_WINT + 4096u*1024u*2u)          // 1024 x 2048 bf16 = 4 MB

__global__ void k_cvt_x(const float* __restrict__ x, __bf16* __restrict__ xb, int n) {
    int i = blockIdx.x * blockDim.x + threadIdx.x;
    if (i < n) xb[i] = (__bf16)x[i];
}

// W: R x C row-major (f32)  ->  WT: C x R row-major (bf16)
__global__ void k_transpose_bf16(const float* __restrict__ W, __bf16* __restrict__ WT,
                                 int R, int C) {
    int i = blockIdx.x * blockDim.x + threadIdx.x;
    if (i >= R * C) return;
    int r = i / C, c = i % C;                 // coalesced read along c
    WT[(size_t)c * R + r] = (__bf16)W[i];
}

// C(MxN,f32) = A(MxK,bf16) * BT(NxK,bf16)^T ; M%256==0, N%64==0, K%64==0
// wave tile: 32(M) x 64(N) = 8 WMMA accumulators; 8 waves stacked in M -> block 256x64
// K-loop is register double-buffered: loads for K+32 issue before WMMAs consume K.
__global__ __launch_bounds__(256) void k_gemm_bf16(
    const __bf16* __restrict__ A, const __bf16* __restrict__ BT,
    float* __restrict__ C, int M, int N, int K)
{
    const int lane = threadIdx.x & 31;
    const int wave = threadIdx.x >> 5;
    const int r    = lane & 15;   // M row (A-lane) / N col (B-lane, C-lane)
    const int half = lane >> 4;   // K-half selector
    const int m0 = blockIdx.x * 256 + wave * 32;
    const int n0 = blockIdx.y * 64;

    const __bf16* aRow0 = A + (size_t)(m0 + r)      * K + half * 16;
    const __bf16* aRow1 = A + (size_t)(m0 + 16 + r) * K + half * 16;
    const __bf16* bRow0 = BT + (size_t)(n0 +  0 + r) * K + half * 16;
    const __bf16* bRow1 = BT + (size_t)(n0 + 16 + r) * K + half * 16;
    const __bf16* bRow2 = BT + (size_t)(n0 + 32 + r) * K + half * 16;
    const __bf16* bRow3 = BT + (size_t)(n0 + 48 + r) * K + half * 16;

    v8f acc[8] = {};
    v16bf a0 = *(const v16bf*)aRow0;
    v16bf a1 = *(const v16bf*)aRow1;
    v16bf b0 = *(const v16bf*)bRow0;
    v16bf b1 = *(const v16bf*)bRow1;
    v16bf b2 = *(const v16bf*)bRow2;
    v16bf b3 = *(const v16bf*)bRow3;

    for (int kb = 32; kb < K; kb += 32) {
        // prefetch next K-block into the second register set (in flight during WMMAs)
        __builtin_prefetch(aRow0 + kb + 32, 0, 3);
        v16bf a0n = *(const v16bf*)(aRow0 + kb);
        v16bf a1n = *(const v16bf*)(aRow1 + kb);
        v16bf b0n = *(const v16bf*)(bRow0 + kb);
        v16bf b1n = *(const v16bf*)(bRow1 + kb);
        v16bf b2n = *(const v16bf*)(bRow2 + kb);
        v16bf b3n = *(const v16bf*)(bRow3 + kb);

        acc[0] = __builtin_amdgcn_wmma_f32_16x16x32_bf16(false, a0, false, b0, (short)0, acc[0], false, false);
        acc[1] = __builtin_amdgcn_wmma_f32_16x16x32_bf16(false, a0, false, b1, (short)0, acc[1], false, false);
        acc[2] = __builtin_amdgcn_wmma_f32_16x16x32_bf16(false, a0, false, b2, (short)0, acc[2], false, false);
        acc[3] = __builtin_amdgcn_wmma_f32_16x16x32_bf16(false, a0, false, b3, (short)0, acc[3], false, false);
        acc[4] = __builtin_amdgcn_wmma_f32_16x16x32_bf16(false, a1, false, b0, (short)0, acc[4], false, false);
        acc[5] = __builtin_amdgcn_wmma_f32_16x16x32_bf16(false, a1, false, b1, (short)0, acc[5], false, false);
        acc[6] = __builtin_amdgcn_wmma_f32_16x16x32_bf16(false, a1, false, b2, (short)0, acc[6], false, false);
        acc[7] = __builtin_amdgcn_wmma_f32_16x16x32_bf16(false, a1, false, b3, (short)0, acc[7], false, false);

        a0 = a0n; a1 = a1n; b0 = b0n; b1 = b1n; b2 = b2n; b3 = b3n;
    }
    // drain last K-block
    acc[0] = __builtin_amdgcn_wmma_f32_16x16x32_bf16(false, a0, false, b0, (short)0, acc[0], false, false);
    acc[1] = __builtin_amdgcn_wmma_f32_16x16x32_bf16(false, a0, false, b1, (short)0, acc[1], false, false);
    acc[2] = __builtin_amdgcn_wmma_f32_16x16x32_bf16(false, a0, false, b2, (short)0, acc[2], false, false);
    acc[3] = __builtin_amdgcn_wmma_f32_16x16x32_bf16(false, a0, false, b3, (short)0, acc[3], false, false);
    acc[4] = __builtin_amdgcn_wmma_f32_16x16x32_bf16(false, a1, false, b0, (short)0, acc[4], false, false);
    acc[5] = __builtin_amdgcn_wmma_f32_16x16x32_bf16(false, a1, false, b1, (short)0, acc[5], false, false);
    acc[6] = __builtin_amdgcn_wmma_f32_16x16x32_bf16(false, a1, false, b2, (short)0, acc[6], false, false);
    acc[7] = __builtin_amdgcn_wmma_f32_16x16x32_bf16(false, a1, false, b3, (short)0, acc[7], false, false);

    // C/D layout: lanes 0-15 hold N=lane, VGPR rr = M row rr (+8 for lanes 16-31)
#pragma unroll
    for (int i = 0; i < 2; ++i)
#pragma unroll
        for (int j = 0; j < 4; ++j)
#pragma unroll
            for (int rr = 0; rr < 8; ++rr)
                C[(size_t)(m0 + i * 16 + rr + 8 * half) * N + n0 + j * 16 + r] = acc[i * 4 + j][rr];
}

// depthwise causal conv-4 + bias + SiLU ; input = first half of xz rows
__global__ void k_conv_silu(const float* __restrict__ xz, const float* __restrict__ cw,
                            const float* __restrict__ cb, float* __restrict__ xc) {
    int i = blockIdx.x * blockDim.x + threadIdx.x;
    if (i >= BL * DI) return;
    int d = i % DI;
    int t = (i / DI) % L_;
    int b = i / (DI * L_);
    const float* xin = xz + (size_t)b * L_ * (2 * DI);
    float s = cb[d];
#pragma unroll
    for (int q = 0; q < DC; ++q) {
        int tt = t - (DC - 1) + q;
        float xv = (tt >= 0) ? xin[(size_t)tt * (2 * DI) + d] : 0.f;
        s += cw[d * DC + q] * xv;
    }
    xc[(size_t)(b * L_ + t) * DI + d] = s / (1.f + __expf(-s));
}

// dbc(BL x 33) = xc(BL x 2048) @ W_x(2048 x 33); j consecutive -> coalesced W_x reads
__global__ void k_dbc(const float* __restrict__ xc, const float* __restrict__ Wx,
                      float* __restrict__ dbc) {
    int i = blockIdx.x * blockDim.x + threadIdx.x;
    if (i >= BL * 33) return;
    int row = i / 33, j = i % 33;
    const float* xr = xc + (size_t)row * DI;
    float acc = 0.f;
    for (int k = 0; k < DI; ++k) acc += xr[k] * Wx[k * 33 + j];
    dbc[(size_t)row * 33 + j] = acc;
}

// selective scan: one thread per (b,d); h[16] in registers;
// dbc rows staged 4 timesteps at a time in LDS (132 floats per fill)
#define TCH 4
__global__ __launch_bounds__(256) void k_scan(
    const float* __restrict__ xz, const float* __restrict__ xc,
    const float* __restrict__ dbc, const float* __restrict__ Alog,
    const float* __restrict__ Dp, const float* __restrict__ Wdt,
    const float* __restrict__ bdt, __bf16* __restrict__ ybf)
{
    __shared__ float sd[TCH * 33];
    const int tid = threadIdx.x;
    const int b = blockIdx.x >> 3;
    const int d = ((blockIdx.x & 7) << 8) + tid;

    float Arow[DS], h[DS];
#pragma unroll
    for (int n = 0; n < DS; ++n) { Arow[n] = -__expf(Alog[d * DS + n]); h[n] = 0.f; }
    const float wdt = Wdt[d], bdv = bdt[d], dpv = Dp[d];

    const float* zb  = xz + (size_t)b * L_ * (2 * DI) + DI + d;
    const float* xcb = xc + (size_t)(b * L_) * DI + d;
    __bf16*      yb  = ybf + (size_t)(b * L_) * DI + d;

    for (int t0 = 0; t0 < L_; t0 += TCH) {
        if (tid < TCH * 33) sd[tid] = dbc[(size_t)(b * L_ + t0) * 33 + tid];
        __syncthreads();
#pragma unroll
        for (int tt = 0; tt < TCH; ++tt) {
            const int t = t0 + tt;
            const float* sr = sd + tt * 33;
            float xcv = xcb[(size_t)t * DI];
            float xdt = sr[0] * wdt + bdv;
            float delta = (xdt > 20.f) ? xdt : log1pf(__expf(xdt));   // softplus
            float du = delta * xcv;
            float y = 0.f;
#pragma unroll
            for (int n = 0; n < DS; ++n) {
                float dA = __expf(delta * Arow[n]);
                h[n] = dA * h[n] + du * sr[1 + n];   // B_sel
                y += h[n] * sr[17 + n];              // C_sel
            }
            y += dpv * xcv;
            float zv = zb[(size_t)t * (2 * DI)];
            y *= zv / (1.f + __expf(-zv));           // * silu(z)
            yb[(size_t)t * DI] = (__bf16)y;
        }
        __syncthreads();
    }
}

extern "C" void kernel_launch(void* const* d_in, const int* in_sizes, int n_in,
                              void* d_out, int out_size, void* d_ws, size_t ws_size,
                              hipStream_t stream) {
    const float* x      = (const float*)d_in[0];
    const float* W_in   = (const float*)d_in[1];
    const float* conv_w = (const float*)d_in[2];
    const float* conv_b = (const float*)d_in[3];
    const float* A_log  = (const float*)d_in[4];
    const float* Dp     = (const float*)d_in[5];
    const float* W_x    = (const float*)d_in[6];
    const float* W_dt   = (const float*)d_in[7];
    const float* b_dt   = (const float*)d_in[8];
    const float* W_out  = (const float*)d_in[9];

    char* ws = (char*)d_ws;
    float*  xz    = (float*)(ws + OFF_XZ);
    float*  xc    = (float*)(ws + OFF_XC);
    float*  dbc   = (float*)(ws + OFF_DBC);
    __bf16* ybf   = (__bf16*)(ws + OFF_YBF);
    __bf16* xbf   = (__bf16*)(ws + OFF_XBF);
    __bf16* WinT  = (__bf16*)(ws + OFF_WINT);
    __bf16* WoutT = (__bf16*)(ws + OFF_WOUTT);

    // prep: bf16 activations + pre-transposed bf16 weights
    k_cvt_x<<<(BL * DM + 255) / 256, 256, 0, stream>>>(x, xbf, BL * DM);
    k_transpose_bf16<<<(DM * 2 * DI + 255) / 256, 256, 0, stream>>>(W_in, WinT, DM, 2 * DI);
    k_transpose_bf16<<<(DI * DM + 255) / 256, 256, 0, stream>>>(W_out, WoutT, DI, DM);

    // xz = x @ W_in   (2048 x 4096, K=1024)
    dim3 g1(BL / 256, (2 * DI) / 64);
    k_gemm_bf16<<<g1, 256, 0, stream>>>(xbf, WinT, xz, BL, 2 * DI, DM);

    k_conv_silu<<<(BL * DI + 255) / 256, 256, 0, stream>>>(xz, conv_w, conv_b, xc);
    k_dbc<<<(BL * 33 + 255) / 256, 256, 0, stream>>>(xc, W_x, dbc);
    k_scan<<<16, 256, 0, stream>>>(xz, xc, dbc, A_log, Dp, W_dt, b_dt, ybf);

    // out = y @ W_out (2048 x 1024, K=2048), f32 output
    dim3 g2(BL / 256, DM / 64);
    k_gemm_bf16<<<g2, 256, 0, stream>>>(ybf, WoutT, (float*)d_out, BL, DM, DI);
}
